// GAT_17617955848390
// MI455X (gfx1250) — matrix operand order repaired and checked by reading, not compiled, not run on hardware
//
#include <hip/hip_runtime.h>

#define NN   4096
#define IND  256
#define OUTD 64
#define NH   8
#define NEG_SLOPE 0.2f
#define LOG2E 1.44269504088896340736f

typedef __attribute__((ext_vector_type(16))) _Float16 v16h;
typedef __attribute__((ext_vector_type(8)))  _Float16 v8h;
typedef __attribute__((ext_vector_type(4)))  _Float16 v4h;
typedef __attribute__((ext_vector_type(8)))  float    v8f;

__device__ __forceinline__ v8f wmma_f16(v16h a, v16h b, v8f c) {
  // D = A(16x32 f16) * B(32x16 f16) + C(16x16 f32)
  return __builtin_amdgcn_wmma_f32_16x16x32_f16(
      /*neg_a=*/false, a, /*neg_b=*/false, b,
      /*c_mod=*/(short)0, c, /*reuse_a=*/false, /*reuse_b=*/false);
}

// ---------------------------------------------------------------------------
// Stage 0a: x (f32, [n][k]) -> x16 (f16, same layout). 4 elems/thread.
// ---------------------------------------------------------------------------
__global__ void __launch_bounds__(256)
cvt_x(const float* __restrict__ x, _Float16* __restrict__ x16) {
  const size_t i = ((size_t)blockIdx.x * 256 + threadIdx.x) * 4;
  float4 v = *(const float4*)(x + i);
  v4h h;
  h[0] = (_Float16)v.x; h[1] = (_Float16)v.y;
  h[2] = (_Float16)v.z; h[3] = (_Float16)v.w;
  *(v4h*)(x16 + i) = h;
}

// ---------------------------------------------------------------------------
// Stage 0b: w (f32, [head][k][o]) -> wT16 (f16, [head][o][k]).
// ---------------------------------------------------------------------------
__global__ void __launch_bounds__(256)
cvt_w(const float* __restrict__ w, _Float16* __restrict__ wT) {
  const int head = blockIdx.y;
  const int idx  = blockIdx.x * 256 + threadIdx.x;  // 0 .. IND*OUTD-1
  const int k = idx >> 6;
  const int o = idx & 63;
  wT[((size_t)head * OUTD + o) * IND + k] =
      (_Float16)w[((size_t)head * IND + k) * OUTD + o];
}

// ---------------------------------------------------------------------------
// Stage 0c: wa = w @ a per head (es/ed folded into stage-1 GEMM as B columns).
// waT layout [head][16][k]: row 0 = w@a_src, row 1 = w@a_dst, rows 2..15 = 0.
// ---------------------------------------------------------------------------
__global__ void __launch_bounds__(256)
cvt_wa(const float* __restrict__ w, const float* __restrict__ a,
       _Float16* __restrict__ waT) {
  const int head = blockIdx.x;
  const int k = threadIdx.x;                       // 0..255
  const float* wp = w + ((size_t)head * IND + k) * OUTD;
  const float* ap = a + (size_t)head * 2 * OUTD;
  float s0 = 0.f, s1 = 0.f;
#pragma unroll 8
  for (int o = 0; o < OUTD; ++o) {
    float hv = wp[o];
    s0 += hv * ap[o];
    s1 += hv * ap[OUTD + o];
  }
  _Float16* base = waT + (size_t)head * 16 * IND + k;
  base[0 * IND] = (_Float16)s0;
  base[1 * IND] = (_Float16)s1;
#pragma unroll
  for (int r = 2; r < 16; ++r) base[r * IND] = (_Float16)0.f;
}

// ---------------------------------------------------------------------------
// Stage 1: h = x @ w per head (f16 WMMA, f32 accum), es/ed via 5th B tile.
//   - writes hT (f16) in [head][o][n] layout (contiguous for stage-2 B loads)
//   - writes es[head][n] = x.(w@a_src), ed[head][n] = x.(w@a_dst)
// One wave (32 threads) per (head, 16-row tile of n).
// ---------------------------------------------------------------------------
__global__ void __launch_bounds__(32)
gat_prep(const _Float16* __restrict__ x16, const _Float16* __restrict__ wT,
         const _Float16* __restrict__ waT, _Float16* __restrict__ hT,
         float* __restrict__ es, float* __restrict__ ed) {
  const int lane = threadIdx.x;     // 0..31
  const int half = lane >> 4;       // which 16-lane half
  const int row  = lane & 15;       // M (rows for A) / N (cols for B,C)
  const int nBase = blockIdx.x * 16;
  const int head  = blockIdx.y;

  v8f acc[4] = {};                  // 4 o-blocks of 16 -> 16x64 tile
  v8f accE = {};                    // cols 0/1 = es/ed

  const _Float16* xrow = x16 + (size_t)(nBase + row) * IND;
  const _Float16* wTh  = wT  + (size_t)head * OUTD * IND + (size_t)row * IND;
  const _Float16* waTh = waT + (size_t)head * 16 * IND + (size_t)row * IND;
  for (int kb = 0; kb < IND; kb += 32) {
    // 16-bit A layout: slots 0..15 -> K=kb+16*half+0..15  (two 16B runs)
    v8h lo = *(const v8h*)(xrow + kb + 16 * half);
    v8h hi = *(const v8h*)(xrow + kb + 16 * half + 8);
    v16h A;
#pragma unroll
    for (int s = 0; s < 8; ++s) { A[s] = lo[s]; A[s + 8] = hi[s]; }

#pragma unroll
    for (int ob = 0; ob < 4; ++ob) {
      // B layout: lane holds column o, K = kb + 16*half + s (contiguous 32B)
      const v16h* bp =
          (const v16h*)(wTh + (size_t)ob * 16 * IND + kb + 16 * half);
      acc[ob] = wmma_f16(A, *bp, acc[ob]);
    }
    accE = wmma_f16(A, *(const v16h*)(waTh + kb + 16 * half), accE);
  }

  // hT[head][o][n] in f16 (contiguous along n for stage-2 B tiles)
#pragma unroll
  for (int ob = 0; ob < 4; ++ob) {
    const int o = ob * 16 + row;
    _Float16* hp = hT + ((size_t)head * OUTD + o) * NN + nBase;
#pragma unroll
    for (int r = 0; r < 8; ++r) hp[r + 8 * half] = (_Float16)acc[ob][r];
  }

  // es (lanes with N==0) / ed (N==1); C layout row M = r + 8*half.
  if (row < 2) {
    float* dst = (row == 0 ? es : ed) + (size_t)head * NN + nBase;
#pragma unroll
    for (int r = 0; r < 8; ++r) dst[r + 8 * half] = accE[r];
  }
}

// ---------------------------------------------------------------------------
// Stage 2: out[n][head][:] = softmax_j(lrelu(es_i + ed_j)) @ h[head][j][:]
// Single pass: row max = lrelu(es_i + max_j ed_j) (lrelu is monotonic).
// Row sums come from a 5th WMMA against an all-ones B (lands in C layout).
// 4 waves/block, each wave owns a 16-row i-tile; block shares LDS-staged ed.
// ---------------------------------------------------------------------------
__global__ void __launch_bounds__(128)
gat_attn(const _Float16* __restrict__ hT, const float* __restrict__ es,
         const float* __restrict__ ed, float* __restrict__ out) {
  __shared__ float lds_ed[NN];
  __shared__ float red[128];

  const int tid  = threadIdx.x;
  const int wave = tid >> 5;
  const int lane = tid & 31;
  const int half = lane >> 4;
  const int row  = lane & 15;
  const int head = blockIdx.y;
  const int iBase = blockIdx.x * 64 + wave * 16;

  // Stage ed row into LDS pre-scaled by log2(e); find per-head max.
  float mx = -3.4e38f;
  const float* edh = ed + (size_t)head * NN;
  for (int j = tid; j < NN; j += 128) {
    float v = edh[j] * LOG2E;
    lds_ed[j] = v;
    mx = fmaxf(mx, v);
  }
  red[tid] = mx;
  __syncthreads();
#pragma unroll
  for (int off = 64; off > 0; off >>= 1) {
    if (tid < off) red[tid] = fmaxf(red[tid], red[tid + off]);
    __syncthreads();
  }
  const float edmax2 = red[0];

  // All in the log2 domain (lrelu commutes with a positive scale).
  const float es2 = es[(size_t)head * NN + iBase + row] * LOG2E;
  const float tm  = es2 + edmax2;
  const float m2  = tm >= 0.f ? tm : NEG_SLOPE * tm;  // exact row max (log2)

  v16h ones;
#pragma unroll
  for (int s = 0; s < 16; ++s) ones[s] = (_Float16)1.0f;

  const _Float16* hTh = hT + (size_t)head * OUTD * NN;
  v8f acc[4] = {};
  v8f accS = {};

#pragma unroll 2
  for (int jb = 0; jb < NN; jb += 32) {
    // Prefetch next chunk's B tiles (gfx1250 global_prefetch_b8).
    if (jb + 32 < NN) {
#pragma unroll
      for (int ob = 0; ob < 4; ++ob)
        __builtin_prefetch(
            hTh + ((size_t)(ob * 16 + row)) * NN + jb + 32 + 16 * half, 0, 3);
    }
    // A tile = probabilities for 16 rows x 32 j (f16 A layout slot->j map)
    float t[16];
    *(float4*)(t + 0)  = *(const float4*)(&lds_ed[jb + 16 * half]);
    *(float4*)(t + 4)  = *(const float4*)(&lds_ed[jb + 16 * half + 4]);
    *(float4*)(t + 8)  = *(const float4*)(&lds_ed[jb + 16 * half + 8]);
    *(float4*)(t + 12) = *(const float4*)(&lds_ed[jb + 16 * half + 12]);
    v16h A;
#pragma unroll
    for (int s = 0; s < 16; ++s) {
      float e = es2 + t[s];
      e = e >= 0.f ? e : NEG_SLOPE * e;                  // leaky relu (log2 dom)
      A[s] = (_Float16)__builtin_amdgcn_exp2f(e - m2);   // p <= 1
    }
    // 4 B tiles: hT[head][o][j] contiguous 16 f16 per lane (32B aligned)
#pragma unroll
    for (int ob = 0; ob < 4; ++ob) {
      const v16h* bp = (const v16h*)(
          hTh + ((size_t)(ob * 16 + row)) * NN + jb + 16 * half);
      acc[ob] = wmma_f16(A, *bp, acc[ob]);
    }
    accS = wmma_f16(A, ones, accS);   // row sums, already in C layout
  }

  // Epilogue: C layout M = r + 8*half, N = row within o-block.
  float inv[8];
#pragma unroll
  for (int r = 0; r < 8; ++r) inv[r] = 1.0f / accS[r];
#pragma unroll
  for (int ob = 0; ob < 4; ++ob) {
    const int o = ob * 16 + row;
#pragma unroll
    for (int r = 0; r < 8; ++r) {
      const int n = iBase + r + 8 * half;
      out[((size_t)n * NH + head) * OUTD + o] = acc[ob][r] * inv[r];
    }
  }
}

extern "C" void kernel_launch(void* const* d_in, const int* in_sizes, int n_in,
                              void* d_out, int out_size, void* d_ws, size_t ws_size,
                              hipStream_t stream) {
  (void)in_sizes; (void)n_in; (void)out_size; (void)ws_size;
  const float* x = (const float*)d_in[0];   // (4096, 256)
  const float* w = (const float*)d_in[1];   // (8, 256, 64)
  const float* a = (const float*)d_in[2];   // (8, 128, 1)
  float* out = (float*)d_out;               // (4096, 8, 64)

  // Workspace layout
  char* ws = (char*)d_ws;
  _Float16* hT  = (_Float16*)ws;                     // 8*64*4096*2  = 4 MB
  ws += (size_t)NH * OUTD * NN * sizeof(_Float16);
  float* es = (float*)ws;                            // 8*4096*4 = 128 KB
  ws += (size_t)NH * NN * sizeof(float);
  float* ed = (float*)ws;                            // 128 KB
  ws += (size_t)NH * NN * sizeof(float);
  _Float16* x16 = (_Float16*)ws;                     // 4096*256*2 = 2 MB
  ws += (size_t)NN * IND * sizeof(_Float16);
  _Float16* wT16 = (_Float16*)ws;                    // 8*64*256*2 = 256 KB
  ws += (size_t)NH * OUTD * IND * sizeof(_Float16);
  _Float16* waT16 = (_Float16*)ws;                   // 8*16*256*2 = 64 KB

  cvt_x<<<dim3((NN * IND) / (256 * 4)), 256, 0, stream>>>(x, x16);
  cvt_w<<<dim3((IND * OUTD) / 256, NH), 256, 0, stream>>>(w, wT16);
  cvt_wa<<<dim3(NH), 256, 0, stream>>>(w, a, waT16);
  gat_prep<<<dim3(NN / 16, NH), 32, 0, stream>>>(x16, wT16, waT16, hT, es, ed);
  gat_attn<<<dim3(NN / 64, NH), 128, 0, stream>>>(hT, es, ed, out);
}
